// PinSAGEModel_7078106104096
// MI455X (gfx1250) — matrix operand order repaired and checked by reading, not compiled, not run on hardware
//
#include <hip/hip_runtime.h>
#include <hip/hip_bf16.h>
#include <cstdint>
#include <cstddef>

// ---------------------------------------------------------------------------
// PinSAGE forward on MI455X (gfx1250, wave32, WMMA f16->f32)
// ---------------------------------------------------------------------------

typedef __attribute__((ext_vector_type(16))) _Float16 v16h;
typedef __attribute__((ext_vector_type(8)))  _Float16 v8h;
typedef __attribute__((ext_vector_type(8)))  float    v8f;

#define N0C 1183744
#define N1C 69632
#define N2C 4096
#define E0C 1114112
#define E1C 65536
#define DH  256

// ---- WMMA fragment helpers (wave32 layouts per CDNA5 ISA 7.12.2) ----------
// B (K x 16): lane n (0..15) holds column nBase+n; K = kBase+{0..7,16..23} (lanes<16), +{8..15,24..31} (lanes>=16).
// BT is stored transposed [N][K] f16, so each lane reads two 8-half runs.
__device__ __forceinline__ v16h b_frag(const _Float16* __restrict__ BT, int ldk,
                                       int nBase, int kBase, int lane) {
  const _Float16* p = BT + (size_t)(nBase + (lane & 15)) * ldk + kBase + ((lane >> 4) << 3);
  v8h lo = *(const v8h*)p;
  v8h hi = *(const v8h*)(p + 16);
  return __builtin_shufflevector(lo, hi, 0, 1, 2, 3, 4, 5, 6, 7, 8, 9, 10, 11, 12, 13, 14, 15);
}

// A (16 x 32) from a row-major f32 row pointer.
// lane<16: row=lane,    K = kBase+{0..7,16..23}
// lane>=16: row=lane-16, K = kBase+{8..15,24..31}   (hiOff = 8)
__device__ __forceinline__ v16h a_frag_g(const float* __restrict__ rowp, int kBase, int hiOff) {
  const float* p = rowp + kBase + hiOff;
  v16h a;
#pragma unroll
  for (int i = 0; i < 8; ++i) { a[i] = (_Float16)p[i]; a[i + 8] = (_Float16)p[i + 16]; }
  return a;
}

__device__ __forceinline__ v16h a_frag_lds(const float* tile, int stride, int kBase, int lane) {
  const float* p = tile + (lane & 15) * stride + kBase + (((lane >> 4) & 1) ? 8 : 0);
  v16h a;
#pragma unroll
  for (int i = 0; i < 8; ++i) { a[i] = (_Float16)p[i]; a[i + 8] = (_Float16)p[i + 16]; }
  return a;
}

__device__ __forceinline__ v8f wmma_step(v16h a, v16h b, v8f c) {
  return __builtin_amdgcn_wmma_f32_16x16x32_f16(false, a, false, b, (short)0, c, false, false);
}

// ---- weight prep: fp32 [K][N] -> f16 transposed [N][K] --------------------
__global__ void prep_weight_kernel(const float* __restrict__ W, _Float16* __restrict__ WT,
                                   int K, int N) {
  int idx = blockIdx.x * blockDim.x + threadIdx.x;
  if (idx >= K * N) return;
  int k = idx % K;
  int n = idx / K;
  WT[idx] = (_Float16)W[(size_t)k * N + n];  // WT[n][k]
}

__global__ void zero_kernel(float* __restrict__ p, size_t n) {
  size_t i = (size_t)blockIdx.x * blockDim.x + threadIdx.x;
  size_t stride = (size_t)gridDim.x * blockDim.x;
  for (; i < n; i += stride) p[i] = 0.f;
}

// ---- fused: h = feats@Wp + bp ; nsrc = relu(h@Q0 + bQ0) -------------------
// One wave per 16-row tile; h tile restaged through LDS (no h round trip to HBM
// for rows >= N1).
__global__ __launch_bounds__(32) void proj_fused_kernel(
    const float* __restrict__ feats,
    const _Float16* __restrict__ WpT, const float* __restrict__ bp,
    const _Float16* __restrict__ QT,  const float* __restrict__ bQ,
    float* __restrict__ h_out, _Float16* __restrict__ nsrc_out) {
  __shared__ float tile[16 * 260];  // stride 260 -> conflict-free banks
  const int lane  = threadIdx.x & 31;
  const int mBase = blockIdx.x * 16;
  const int col   = lane & 15;
  const int hiOff = (lane >> 4) ? 8 : 0;

  const float* rowp = feats + (size_t)(mBase + col) * 128;
  if (mBase + 16 < N0C)
    __builtin_prefetch(feats + (size_t)(mBase + 16 + col) * 128, 0, 1);

  v16h a[4];
#pragma unroll
  for (int j = 0; j < 4; ++j) a[j] = a_frag_g(rowp, 32 * j, hiOff);

  v8f acc[16];
#pragma unroll
  for (int t = 0; t < 16; ++t) {
    v8f c = {};
#pragma unroll
    for (int j = 0; j < 4; ++j)
      c = wmma_step(a[j], b_frag(WpT, 128, t * 16, 32 * j, lane), c);
    float bv = bp[t * 16 + col];
#pragma unroll
    for (int r = 0; r < 8; ++r) c[r] += bv;
    acc[t] = c;
  }

  if (mBase < N1C) {  // N1 is a multiple of 16 -> whole tile in range
    float* hp = h_out + (size_t)mBase * DH;
#pragma unroll
    for (int t = 0; t < 16; ++t)
#pragma unroll
      for (int r = 0; r < 8; ++r)
        hp[(size_t)(hiOff + r) * DH + t * 16 + col] = acc[t][r];
  }
#pragma unroll
  for (int t = 0; t < 16; ++t)
#pragma unroll
    for (int r = 0; r < 8; ++r)
      tile[(hiOff + r) * 260 + t * 16 + col] = acc[t][r];
  __syncthreads();

  v16h a2[8];
#pragma unroll
  for (int j = 0; j < 8; ++j) a2[j] = a_frag_lds(tile, 260, 32 * j, lane);

  _Float16* np = nsrc_out + (size_t)mBase * DH;
#pragma unroll
  for (int t = 0; t < 16; ++t) {
    v8f c = {};
#pragma unroll
    for (int j = 0; j < 8; ++j)
      c = wmma_step(a2[j], b_frag(QT, 256, t * 16, 32 * j, lane), c);
    float bv = bQ[t * 16 + col];
#pragma unroll
    for (int r = 0; r < 8; ++r) {
      float z = fmaxf(c[r] + bv, 0.f);
      np[(size_t)(hiOff + r) * DH + t * 16 + col] = (_Float16)z;
    }
  }
}

// ---- weighted message scatter: agg[dst] += w * nsrc[src] ------------------
// One wave per edge; destination rows (<=71MB) are L2-resident -> fast atomics.
__global__ __launch_bounds__(256) void edge_scatter_kernel(
    const _Float16* __restrict__ nsrc, const int* __restrict__ src,
    const int* __restrict__ dst, const float* __restrict__ w,
    float* __restrict__ agg, float* __restrict__ wsum, int E) {
  int gid  = blockIdx.x * blockDim.x + threadIdx.x;
  int e    = gid >> 5;
  int lane = gid & 31;
  if (e >= E) return;
  int   s  = src[e];
  int   d  = dst[e];
  float wv = w[e];
  const _Float16* p = nsrc + (size_t)s * DH + lane * 8;
  float*          q = agg  + (size_t)d * DH + lane * 8;
  v8h hv = *(const v8h*)p;
#pragma unroll
  for (int i = 0; i < 8; ++i) atomicAdd(q + i, wv * (float)hv[i]);
  if (lane == 0) atomicAdd(wsum + d, wv);
}

// ---- shared combine body: z = relu([agg/ws | hdst] @ W + bW); z /= ||z|| --
__device__ __forceinline__ void combine_core(
    const float* __restrict__ agg, const float* __restrict__ wsum,
    const float* __restrict__ hdst,
    const _Float16* __restrict__ WT, const float* __restrict__ bW,
    float* tile, int mBase, int lane, v8f acc[16]) {
  const int col   = lane & 15;
  const int half  = lane >> 4;
  // stage A = [agg * 1/max(ws,1) | hdst] (16 x 512 f32) into LDS
  {
    const int grow = mBase + col;
    const float* sp;
    float scale;
    if (half == 0) { sp = agg  + (size_t)grow * DH; scale = 1.f / fmaxf(wsum[grow], 1.f); }
    else           { sp = hdst + (size_t)grow * DH; scale = 1.f; }
    float* dp = tile + col * 516 + half * 256;
#pragma unroll 4
    for (int c = 0; c < 256; c += 4) {
      float4 v = *(const float4*)(sp + c);
      dp[c + 0] = v.x * scale; dp[c + 1] = v.y * scale;
      dp[c + 2] = v.z * scale; dp[c + 3] = v.w * scale;
    }
  }
  __syncthreads();

#pragma unroll
  for (int t = 0; t < 16; ++t) {
    v8f c = {};
#pragma unroll
    for (int j = 0; j < 16; ++j) {
      v16h av = a_frag_lds(tile, 516, 32 * j, lane);
      c = wmma_step(av, b_frag(WT, 512, t * 16, 32 * j, lane), c);
    }
    float bv = bW[t * 16 + col];
#pragma unroll
    for (int r = 0; r < 8; ++r) c[r] = fmaxf(c[r] + bv, 0.f);
    acc[t] = c;
  }

  // row L2 norm: rows r (lanes 0-15) / r+8 (lanes 16-31); reduce within halves
  float sq[8];
#pragma unroll
  for (int r = 0; r < 8; ++r) sq[r] = 0.f;
#pragma unroll
  for (int t = 0; t < 16; ++t)
#pragma unroll
    for (int r = 0; r < 8; ++r) sq[r] += acc[t][r] * acc[t][r];
#pragma unroll
  for (int m = 1; m < 16; m <<= 1)
#pragma unroll
    for (int r = 0; r < 8; ++r) sq[r] += __shfl_xor(sq[r], m, 32);
#pragma unroll
  for (int r = 0; r < 8; ++r) {
    float nn = sqrtf(sq[r]);
    float inv = (nn == 0.f) ? 1.f : (1.f / nn);
#pragma unroll
    for (int t = 0; t < 16; ++t) acc[t][r] *= inv;
  }
}

// layer-1 combine, fused with layer-2 source projection Q1
__global__ __launch_bounds__(32) void combine1_kernel(
    const float* __restrict__ agg, const float* __restrict__ wsum,
    const float* __restrict__ hdst,
    const _Float16* __restrict__ WT, const float* __restrict__ bW,
    const _Float16* __restrict__ QT, const float* __restrict__ bQ,
    float* __restrict__ h_store, int hStoreRows,
    _Float16* __restrict__ nsrc_out) {
  __shared__ float tile[16 * 516];
  const int lane  = threadIdx.x & 31;
  const int mBase = blockIdx.x * 16;
  const int col   = lane & 15;
  const int hiOff = (lane >> 4) ? 8 : 0;

  v8f acc[16];
  combine_core(agg, wsum, hdst, WT, bW, tile, mBase, lane, acc);

  if (mBase < hStoreRows) {
    float* hp = h_store + (size_t)mBase * DH;
#pragma unroll
    for (int t = 0; t < 16; ++t)
#pragma unroll
      for (int r = 0; r < 8; ++r)
        hp[(size_t)(hiOff + r) * DH + t * 16 + col] = acc[t][r];
  }

  // restage normalized tile, apply Q (next layer's source projection)
  __syncthreads();
#pragma unroll
  for (int t = 0; t < 16; ++t)
#pragma unroll
    for (int r = 0; r < 8; ++r)
      tile[(hiOff + r) * 516 + t * 16 + col] = acc[t][r];
  __syncthreads();

  v16h a2[8];
#pragma unroll
  for (int j = 0; j < 8; ++j) a2[j] = a_frag_lds(tile, 516, 32 * j, lane);

  _Float16* np = nsrc_out + (size_t)mBase * DH;
#pragma unroll
  for (int t = 0; t < 16; ++t) {
    v8f c = {};
#pragma unroll
    for (int j = 0; j < 8; ++j)
      c = wmma_step(a2[j], b_frag(QT, 256, t * 16, 32 * j, lane), c);
    float bv = bQ[t * 16 + col];
#pragma unroll
    for (int r = 0; r < 8; ++r) {
      float z = fmaxf(c[r] + bv, 0.f);
      np[(size_t)(hiOff + r) * DH + t * 16 + col] = (_Float16)z;
    }
  }
}

// layer-2 combine, fused residual: out = resid + h2
__global__ __launch_bounds__(32) void combine2_kernel(
    const float* __restrict__ agg, const float* __restrict__ wsum,
    const float* __restrict__ hdst,
    const _Float16* __restrict__ WT, const float* __restrict__ bW,
    const float* __restrict__ resid, float* __restrict__ out) {
  __shared__ float tile[16 * 516];
  const int lane  = threadIdx.x & 31;
  const int mBase = blockIdx.x * 16;
  const int col   = lane & 15;
  const int hiOff = (lane >> 4) ? 8 : 0;

  v8f acc[16];
  combine_core(agg, wsum, hdst, WT, bW, tile, mBase, lane, acc);

  const float* rp = resid + (size_t)mBase * DH;
  float*       op = out   + (size_t)mBase * DH;
#pragma unroll
  for (int t = 0; t < 16; ++t)
#pragma unroll
    for (int r = 0; r < 8; ++r) {
      size_t o = (size_t)(hiOff + r) * DH + t * 16 + col;
      op[o] = acc[t][r] + rp[o];
    }
}

// ---------------------------------------------------------------------------
extern "C" void kernel_launch(void* const* d_in, const int* in_sizes, int n_in,
                              void* d_out, int out_size, void* d_ws, size_t ws_size,
                              hipStream_t stream) {
  (void)in_sizes; (void)n_in; (void)out_size; (void)ws_size;
  const float* feats = (const float*)d_in[0];
  const float* w0    = (const float*)d_in[1];
  const float* w1    = (const float*)d_in[2];
  const float* Wp    = (const float*)d_in[3];
  const float* bp    = (const float*)d_in[4];
  const float* Q0    = (const float*)d_in[5];
  const float* bQ0   = (const float*)d_in[6];
  const float* W0    = (const float*)d_in[7];
  const float* bW0   = (const float*)d_in[8];
  const float* Q1    = (const float*)d_in[9];
  const float* bQ1   = (const float*)d_in[10];
  const float* W1    = (const float*)d_in[11];
  const float* bW1   = (const float*)d_in[12];
  const int*   src0  = (const int*)d_in[13];
  const int*   dst0  = (const int*)d_in[14];
  const int*   src1  = (const int*)d_in[15];
  const int*   dst1  = (const int*)d_in[16];
  float* out = (float*)d_out;

  // workspace layout (256B aligned)
  char*  base = (char*)d_ws;
  size_t off  = 0;
  auto take = [&](size_t bytes) -> void* {
    void* p = base + off;
    off += (bytes + 255) & ~(size_t)255;
    return p;
  };
  _Float16* WpT   = (_Float16*)take((size_t)256 * 128 * 2);
  _Float16* Q0T   = (_Float16*)take((size_t)256 * 256 * 2);
  _Float16* W0T   = (_Float16*)take((size_t)256 * 512 * 2);
  _Float16* Q1T   = (_Float16*)take((size_t)256 * 256 * 2);
  _Float16* W1T   = (_Float16*)take((size_t)256 * 512 * 2);
  float*    h0    = (float*)   take((size_t)N1C * DH * 4);
  _Float16* nsrc0 = (_Float16*)take((size_t)N0C * DH * 2);
  float*    agg0  = (float*)   take((size_t)N1C * DH * 4);
  float*    wsum0 = (float*)   take((size_t)N1C * 4);
  float*    h1    = (float*)   take((size_t)N2C * DH * 4);
  _Float16* nsrc1 = (_Float16*)take((size_t)N1C * DH * 2);
  float*    agg1  = (float*)   take((size_t)N2C * DH * 4);
  float*    wsum1 = (float*)   take((size_t)N2C * 4);

  // 1) weight prep (fp32 -> f16 transposed)
  auto prep = [&](const float* W, _Float16* WT, int K, int N) {
    int n = K * N;
    prep_weight_kernel<<<(n + 255) / 256, 256, 0, stream>>>(W, WT, K, N);
  };
  prep(Wp, WpT, 128, 256);
  prep(Q0, Q0T, 256, 256);
  prep(W0, W0T, 512, 256);
  prep(Q1, Q1T, 256, 256);
  prep(W1, W1T, 512, 256);

  // 2) zero accumulators (ws is poisoned, and graph replays need fresh zeros)
  zero_kernel<<<4096, 256, 0, stream>>>(agg0,  (size_t)N1C * DH);
  zero_kernel<<<256,  256, 0, stream>>>(wsum0, (size_t)N1C);
  zero_kernel<<<1024, 256, 0, stream>>>(agg1,  (size_t)N2C * DH);
  zero_kernel<<<16,   256, 0, stream>>>(wsum1, (size_t)N2C);

  // 3) h = feats@Wp+bp (store rows<N1) fused with nsrc0 = relu(h@Q0+bQ0)
  proj_fused_kernel<<<N0C / 16, 32, 0, stream>>>(feats, WpT, bp, Q0T, bQ0, h0, nsrc0);

  // 4) layer-1 weighted scatter
  edge_scatter_kernel<<<(E0C * 32) / 256, 256, 0, stream>>>(nsrc0, src0, dst0, w0,
                                                            agg0, wsum0, E0C);

  // 5) layer-1 combine (h1 rows<N2) fused with nsrc1 = relu(h1@Q1+bQ1)
  combine1_kernel<<<N1C / 16, 32, 0, stream>>>(agg0, wsum0, h0, W0T, bW0,
                                               Q1T, bQ1, h1, N2C, nsrc1);

  // 6) layer-2 weighted scatter
  edge_scatter_kernel<<<(E1C * 32) / 256, 256, 0, stream>>>(nsrc1, src1, dst1, w1,
                                                            agg1, wsum1, E1C);

  // 7) layer-2 combine + residual -> out
  combine2_kernel<<<N2C / 16, 32, 0, stream>>>(agg1, wsum1, h1, W1T, bW1, h0, out);
}
